// DynamicGNN_13365938226088
// MI455X (gfx1250) — matrix operand order repaired and checked by reading, not compiled, not run on hardware
//
#include <hip/hip_runtime.h>
#include <hip/hip_bf16.h>

#define N_NODES 40000
#define N_EDGES 640000
#define ETOT    (N_EDGES + N_NODES)
#define F_INC   128
#define HIDC    64
#define NHEADS  4
#define NGRAPH  64

typedef __attribute__((ext_vector_type(16))) _Float16 v16h;
typedef __attribute__((ext_vector_type(8)))  float    v8f;

// ---------------- utility ----------------
__global__ void zero_f32_k(float* __restrict__ p, size_t n) {
  size_t i = (size_t)blockIdx.x * blockDim.x + threadIdx.x;
  if (i < n) p[i] = 0.0f;
}

// order-preserving float<->uint encoding for atomic segment-max
__device__ __forceinline__ unsigned enc_ord(float f) {
  unsigned u = __float_as_uint(f);
  return (u & 0x80000000u) ? ~u : (u | 0x80000000u);
}
__device__ __forceinline__ float dec_ord(unsigned u) {
  u = (u & 0x80000000u) ? (u & 0x7fffffffu) : ~u;
  return __uint_as_float(u);
}

// ---------------- WMMA fragment packing ----------------
// A [M,K] f32 row-major -> per-wave fragments [mtile][kstep][lane][8 dwords]
// (ISA 7.12.2: 16-bit A 16x32 layout; lanes 0-15 rows M, K pairs 0..7/16..23; lanes 16-31 +8)
__global__ void pack_a_k(const float* __restrict__ A, _Float16* __restrict__ out,
                         int mtiles, int ksteps, int K) {
  size_t idx = (size_t)blockIdx.x * blockDim.x + threadIdx.x;
  size_t total = (size_t)mtiles * ksteps * 256;
  if (idx >= total) return;
  int v    = (int)(idx & 7);
  int lane = (int)((idx >> 3) & 31);
  size_t rest = idx >> 8;
  int ks = (int)(rest % ksteps);
  int mt = (int)(rest / ksteps);
  int m  = mt * 16 + (lane & 15);
  int k0 = ks * 32 + ((v < 4) ? (2 * v) : (16 + 2 * (v - 4))) + ((lane < 16) ? 0 : 8);
  const float* row = A + (size_t)m * K + k0;
  out[idx * 2]     = (_Float16)row[0];
  out[idx * 2 + 1] = (_Float16)row[1];
}

// B [K,N] f32 row-major -> fragments [ntile][kstep][lane][8 dwords]
// (lanes 0-15 hold K pairs 0..7/16..23, lanes 16-31 +8; lane%16 = column)
__global__ void pack_b_k(const float* __restrict__ B, _Float16* __restrict__ out,
                         int ntiles, int ksteps, int ldb) {
  size_t idx = (size_t)blockIdx.x * blockDim.x + threadIdx.x;
  size_t total = (size_t)ntiles * ksteps * 256;
  if (idx >= total) return;
  int v    = (int)(idx & 7);
  int lane = (int)((idx >> 3) & 31);
  size_t rest = idx >> 8;
  int ks = (int)(rest % ksteps);
  int nt = (int)(rest / ksteps);
  int n  = nt * 16 + (lane & 15);
  int k0 = ks * 32 + ((v < 4) ? (2 * v) : (16 + 2 * (v - 4))) + ((lane < 16) ? 0 : 8);
  out[idx * 2]     = (_Float16)B[(size_t)k0 * ldb + n];
  out[idx * 2 + 1] = (_Float16)B[(size_t)(k0 + 1) * ldb + n];
}

// ---------------- WMMA GEMM: one 16x16 f32 tile per wave ----------------
// KSTEPS is a template parameter so the two-deep fragment ping-pong fully
// unrolls: buffer indices and tail conditions are compile-time constants,
// loads for step ks+1 issue before the WMMA for step ks (partial loadcnt
// waits, no register rotation moves).
template <int KSTEPS>
__global__ void gemm_wmma_k(const _Float16* __restrict__ Ap, const _Float16* __restrict__ Bp,
                            float* __restrict__ C, int mtiles, int ntiles, int ldc) {
  int wv   = (int)((blockIdx.x * blockDim.x + threadIdx.x) >> 5);
  int lane = threadIdx.x & 31;
  if (wv >= mtiles * ntiles) return;
  int mt = wv / ntiles, nt = wv % ntiles;
  const v16h* av = (const v16h*)Ap + (size_t)mt * KSTEPS * 32 + lane;
  const v16h* bv = (const v16h*)Bp + (size_t)nt * KSTEPS * 32 + lane;
  v8f c = {};
  v16h a[2], b[2];
  a[0] = av[0];
  b[0] = bv[0];
#pragma unroll
  for (int ks = 0; ks < KSTEPS; ++ks) {
    const int cur = ks & 1, nxt = cur ^ 1;
    if (ks + 1 < KSTEPS) {            // compile-time resolved per unrolled copy
      a[nxt] = av[(size_t)(ks + 1) * 32];
      b[nxt] = bv[(size_t)(ks + 1) * 32];
    }
    if (ks + 2 < KSTEPS) __builtin_prefetch(av + (size_t)(ks + 2) * 32, 0, 1);
    c = __builtin_amdgcn_wmma_f32_16x16x32_f16(false, a[cur], false, b[cur],
                                               (short)0, c, false, false);
  }
  int col  = nt * 16 + (lane & 15);
  int rowb = mt * 16 + ((lane < 16) ? 0 : 8);
#pragma unroll
  for (int r = 0; r < 8; ++r) C[(size_t)(rowb + r) * ldc + col] = c[r];
}

// ---------------- GAT attention pieces ----------------
__global__ void att_logits_k(const float* __restrict__ h, const float* __restrict__ asrc,
                             const float* __restrict__ adst, float* __restrict__ a_s,
                             float* __restrict__ a_d, int H, int C) {
  int tid = blockIdx.x * blockDim.x + threadIdx.x;
  if (tid >= N_NODES * H) return;
  int n = tid / H, hh = tid % H;
  const float* hp = h + (size_t)n * H * C + (size_t)hh * C;
  const float* sp = asrc + hh * C;
  const float* dp = adst + hh * C;
  float ss = 0.f, sd = 0.f;
  for (int c = 0; c < C; ++c) { float xv = hp[c]; ss += xv * sp[c]; sd += xv * dp[c]; }
  a_s[tid] = ss; a_d[tid] = sd;
}

__global__ void edge_logit_max_k(const float* __restrict__ a_s, const float* __restrict__ a_d,
                                 const int* __restrict__ src, const int* __restrict__ dst,
                                 float* __restrict__ e_out, unsigned* __restrict__ m_enc, int H) {
  int tid = blockIdx.x * blockDim.x + threadIdx.x;
  if (tid >= ETOT * H) return;
  int eid = tid / H, hh = tid % H;
  int s, d;
  if (eid < N_EDGES) { s = src[eid]; d = dst[eid]; } else { s = d = eid - N_EDGES; }
  float e = a_s[s * H + hh] + a_d[d * H + hh];
  e = (e > 0.f) ? e : 0.2f * e;              // leaky_relu(.,0.2)
  e_out[tid] = e;
  atomicMax(&m_enc[d * H + hh], enc_ord(e));
}

__global__ void edge_exp_sum_k(const int* __restrict__ src, const int* __restrict__ dst,
                               const unsigned* __restrict__ m_enc, float* __restrict__ e_buf,
                               float* __restrict__ denom, int H) {
  int tid = blockIdx.x * blockDim.x + threadIdx.x;
  if (tid >= ETOT * H) return;
  int eid = tid / H, hh = tid % H;
  int d = (eid < N_EDGES) ? dst[eid] : (eid - N_EDGES);
  float ex = expf(e_buf[tid] - dec_ord(m_enc[d * H + hh]));
  e_buf[tid] = ex;
  atomicAdd(&denom[d * H + hh], ex);
}

__global__ void edge_scatter_k(const int* __restrict__ src, const int* __restrict__ dst,
                               const float* __restrict__ ex, const float* __restrict__ denom,
                               const float* __restrict__ h, float* __restrict__ out,
                               int H, int C) {
  long long tid   = (long long)blockIdx.x * blockDim.x + threadIdx.x;
  long long total = (long long)ETOT * H * C;
  if (tid >= total) return;
  int hc  = H * C;
  int eid = (int)(tid / hc);
  int t   = (int)(tid % hc);
  int hh  = t / C;
  int s, d;
  if (eid < N_EDGES) { s = src[eid]; d = dst[eid]; } else { s = d = eid - N_EDGES; }
  float alpha = ex[(size_t)eid * H + hh] / denom[d * H + hh];
  atomicAdd(&out[(size_t)d * hc + t], alpha * h[(size_t)s * hc + t]);
}

// ---------------- bias + LayerNorm + ReLU (one node per block, blockDim=D) ----------------
__global__ void bias_ln_relu_k(const float* __restrict__ in, const float* __restrict__ bias,
                               const float* __restrict__ g, const float* __restrict__ be,
                               float* __restrict__ out, int D) {
  extern __shared__ float sm[];
  int n = blockIdx.x, c = threadIdx.x;
  float x = in[(size_t)n * D + c] + bias[c];
  sm[c] = x; __syncthreads();
  for (int s2 = blockDim.x >> 1; s2 > 0; s2 >>= 1) { if (c < s2) sm[c] += sm[c + s2]; __syncthreads(); }
  float mu = sm[0] / (float)D;
  __syncthreads();
  float d0 = x - mu;
  sm[c] = d0 * d0; __syncthreads();
  for (int s2 = blockDim.x >> 1; s2 > 0; s2 >>= 1) { if (c < s2) sm[c] += sm[c + s2]; __syncthreads(); }
  float var = sm[0] / (float)D;
  float y = d0 * rsqrtf(var + 1e-5f) * g[c] + be[c];
  out[(size_t)n * D + c] = fmaxf(y, 0.f);
}

// ---------------- pooling + head ----------------
__global__ void pool_sum_k(const float* __restrict__ h, const int* __restrict__ batch,
                           float* __restrict__ sums, float* __restrict__ cnt) {
  int tid = blockIdx.x * blockDim.x + threadIdx.x;
  if (tid >= N_NODES * HIDC) return;
  int n = tid / HIDC, c = tid % HIDC;
  int b = batch[n];
  atomicAdd(&sums[b * HIDC + c], h[(size_t)n * HIDC + c]);
  if (c == 0) atomicAdd(&cnt[b], 1.0f);
}

__global__ void final_out_k(const float* __restrict__ sums, const float* __restrict__ cnt,
                            const float* __restrict__ Wo, const float* __restrict__ bo,
                            float* __restrict__ out) {
  int g = threadIdx.x;
  if (g >= NGRAPH) return;
  float cn = fmaxf(cnt[g], 1.0f);
  float acc = bo[0];
  for (int c = 0; c < HIDC; ++c) acc += (sums[g * HIDC + c] / cn) * Wo[c];
  out[g] = acc;
}

// ---------------- host launcher ----------------
extern "C" void kernel_launch(void* const* d_in, const int* in_sizes, int n_in,
                              void* d_out, int out_size, void* d_ws, size_t ws_size,
                              hipStream_t stream) {
  (void)in_sizes; (void)n_in; (void)out_size; (void)ws_size;
  const float* x   = (const float*)d_in[0];
  const int*   ei  = (const int*)d_in[1];
  const int*   bat = (const int*)d_in[2];
  const float* W1  = (const float*)d_in[3];
  const float* as1 = (const float*)d_in[4];
  const float* ad1 = (const float*)d_in[5];
  const float* b1  = (const float*)d_in[6];
  const float* g1  = (const float*)d_in[7];
  const float* be1 = (const float*)d_in[8];
  const float* W2  = (const float*)d_in[9];
  const float* as2 = (const float*)d_in[10];
  const float* ad2 = (const float*)d_in[11];
  const float* b2  = (const float*)d_in[12];
  const float* g2  = (const float*)d_in[13];
  const float* be2 = (const float*)d_in[14];
  const float* Wo  = (const float*)d_in[15];
  const float* bo  = (const float*)d_in[16];
  float* out = (float*)d_out;

  const int* src = ei;
  const int* dst = ei + N_EDGES;

  char* w = (char*)d_ws;
  auto alloc = [&](size_t bytes) -> void* {
    void* p = (void*)w;
    w += (bytes + 255) & ~(size_t)255;
    return p;
  };
  float*    h1    = (float*)alloc((size_t)N_NODES * 256 * 4);   // h = x@W1; reused as LN1 output
  float*    out1  = (float*)alloc((size_t)N_NODES * 256 * 4);   // GAT1 aggregation
  _Float16* Apack = (_Float16*)alloc((size_t)N_NODES * 256 * 2); // A fragments (both layers)
  _Float16* Bpack = (_Float16*)alloc((size_t)256 * 256 * 2);     // B fragments
  float*    ebuf  = (float*)alloc((size_t)ETOT * NHEADS * 4);    // edge logits / exp (both layers)
  float*    a_s1  = (float*)alloc((size_t)N_NODES * NHEADS * 4);
  float*    a_d1  = (float*)alloc((size_t)N_NODES * NHEADS * 4);
  unsigned* m1    = (unsigned*)alloc((size_t)N_NODES * NHEADS * 4);
  float*    dn1   = (float*)alloc((size_t)N_NODES * NHEADS * 4);
  float*    h2    = (float*)alloc((size_t)N_NODES * HIDC * 4);   // reused as LN2 output
  float*    out2  = (float*)alloc((size_t)N_NODES * HIDC * 4);
  float*    a_s2  = (float*)alloc((size_t)N_NODES * 4);
  float*    a_d2  = (float*)alloc((size_t)N_NODES * 4);
  unsigned* m2    = (unsigned*)alloc((size_t)N_NODES * 4);
  float*    dn2   = (float*)alloc((size_t)N_NODES * 4);
  float*    sums  = (float*)alloc((size_t)NGRAPH * HIDC * 4);
  float*    cnt   = (float*)alloc((size_t)NGRAPH * 4);

  auto gridFor = [](size_t n, int b) { return (unsigned)((n + (size_t)b - 1) / (size_t)b); };

  // re-zero accumulators every call (graph-replay safe)
  zero_f32_k<<<gridFor((size_t)N_NODES * 256, 256), 256, 0, stream>>>(out1, (size_t)N_NODES * 256);
  zero_f32_k<<<gridFor((size_t)N_NODES * NHEADS, 256), 256, 0, stream>>>((float*)m1, (size_t)N_NODES * NHEADS);
  zero_f32_k<<<gridFor((size_t)N_NODES * NHEADS, 256), 256, 0, stream>>>(dn1, (size_t)N_NODES * NHEADS);
  zero_f32_k<<<gridFor((size_t)N_NODES * HIDC, 256), 256, 0, stream>>>(out2, (size_t)N_NODES * HIDC);
  zero_f32_k<<<gridFor((size_t)N_NODES, 256), 256, 0, stream>>>((float*)m2, (size_t)N_NODES);
  zero_f32_k<<<gridFor((size_t)N_NODES, 256), 256, 0, stream>>>(dn2, (size_t)N_NODES);
  zero_f32_k<<<gridFor((size_t)NGRAPH * HIDC, 256), 256, 0, stream>>>(sums, (size_t)NGRAPH * HIDC);
  zero_f32_k<<<1, 64, 0, stream>>>(cnt, (size_t)NGRAPH);

  // ---- layer 1: h1 = x @ W1 (M=40000,K=128,N=256) via WMMA ----
  {
    const int mt = N_NODES / 16, ks = F_INC / 32, nt = 256 / 16;
    pack_a_k<<<gridFor((size_t)mt * ks * 256, 256), 256, 0, stream>>>(x, Apack, mt, ks, F_INC);
    pack_b_k<<<gridFor((size_t)nt * ks * 256, 256), 256, 0, stream>>>(W1, Bpack, nt, ks, 256);
    gemm_wmma_k<F_INC / 32><<<gridFor((size_t)mt * nt * 32, 256), 256, 0, stream>>>(Apack, Bpack, h1, mt, nt, 256);
  }
  att_logits_k<<<gridFor((size_t)N_NODES * NHEADS, 256), 256, 0, stream>>>(h1, as1, ad1, a_s1, a_d1, NHEADS, HIDC);
  edge_logit_max_k<<<gridFor((size_t)ETOT * NHEADS, 256), 256, 0, stream>>>(a_s1, a_d1, src, dst, ebuf, m1, NHEADS);
  edge_exp_sum_k<<<gridFor((size_t)ETOT * NHEADS, 256), 256, 0, stream>>>(src, dst, m1, ebuf, dn1, NHEADS);
  edge_scatter_k<<<gridFor((size_t)ETOT * NHEADS * HIDC, 256), 256, 0, stream>>>(src, dst, ebuf, dn1, h1, out1, NHEADS, HIDC);
  bias_ln_relu_k<<<N_NODES, 256, 256 * 4, stream>>>(out1, b1, g1, be1, h1, 256);

  // ---- layer 2: h2 = h1n @ W2 (M=40000,K=256,N=64) via WMMA ----
  {
    const int mt = N_NODES / 16, ks = 256 / 32, nt = HIDC / 16;
    pack_a_k<<<gridFor((size_t)mt * ks * 256, 256), 256, 0, stream>>>(h1, Apack, mt, ks, 256);
    pack_b_k<<<gridFor((size_t)nt * ks * 256, 256), 256, 0, stream>>>(W2, Bpack, nt, ks, HIDC);
    gemm_wmma_k<256 / 32><<<gridFor((size_t)mt * nt * 32, 256), 256, 0, stream>>>(Apack, Bpack, h2, mt, nt, HIDC);
  }
  att_logits_k<<<gridFor((size_t)N_NODES, 256), 256, 0, stream>>>(h2, as2, ad2, a_s2, a_d2, 1, HIDC);
  edge_logit_max_k<<<gridFor((size_t)ETOT, 256), 256, 0, stream>>>(a_s2, a_d2, src, dst, ebuf, m2, 1);
  edge_exp_sum_k<<<gridFor((size_t)ETOT, 256), 256, 0, stream>>>(src, dst, m2, ebuf, dn2, 1);
  edge_scatter_k<<<gridFor((size_t)ETOT * HIDC, 256), 256, 0, stream>>>(src, dst, ebuf, dn2, h2, out2, 1, HIDC);
  bias_ln_relu_k<<<N_NODES, HIDC, HIDC * 4, stream>>>(out2, b2, g2, be2, h2, HIDC);

  // ---- pooling + linear head ----
  pool_sum_k<<<gridFor((size_t)N_NODES * HIDC, 256), 256, 0, stream>>>(h2, bat, sums, cnt);
  final_out_k<<<1, 64, 0, stream>>>(sums, cnt, Wo, bo, out);
}
